// ReadwiseHyena_87729001988820
// MI455X (gfx1250) — compile-verified, matrix operand order checked by reading, not run on hardware
//
#include <hip/hip_runtime.h>

typedef _Float16 h16;
typedef __attribute__((ext_vector_type(16))) _Float16 v16h;
typedef __attribute__((ext_vector_type(8)))  _Float16 v8h;
typedef __attribute__((ext_vector_type(8)))  float    v8f;
typedef __attribute__((ext_vector_type(4)))  unsigned int v4u;
typedef __attribute__((ext_vector_type(8)))  unsigned int v8u;

#define DIM 128
#define RL  128

// f16 workspace layout (offsets in halfs): transposed weights, row = output col, 128 contiguous K
#define WS_WPROJT 0        // [384][128]
#define WS_W1T    49152    // [128][128]
#define WS_W2T    65536    // [256][128]
#define WS_WOUTT  98304    // [128][128]
#define WS_TOTAL  114688

// ---------------- weight prep: fp32 -> f16, transposed ----------------
__global__ __launch_bounds__(256) void prep_weights(
    const float* __restrict__ Wproj, const float* __restrict__ W1,
    const float* __restrict__ W2, const float* __restrict__ Wout,
    h16* __restrict__ wf) {
  int tid = blockIdx.x * blockDim.x + threadIdx.x;
  if (tid < 49152) {                       // WprojT[c][k] = Wproj[k][c], c<384
    int c = tid >> 7, k = tid & 127;
    wf[WS_WPROJT + tid] = (h16)Wproj[k * 384 + c];
  } else if (tid < 65536) {                // W1T[n][k]
    int t = tid - 49152; int n = t >> 7, k = t & 127;
    wf[WS_W1T + t] = (h16)W1[k * 128 + n];
  } else if (tid < 98304) {                // W2T[n][k], n<256
    int t = tid - 65536; int n = t >> 7, k = t & 127;
    wf[WS_W2T + t] = (h16)W2[k * 256 + n];
  } else if (tid < WS_TOTAL) {             // WoutT[n][k]
    int t = tid - 98304; int n = t >> 7, k = t & 127;
    wf[WS_WOUTT + t] = (h16)Wout[k * 128 + n];
  }
}

// ---------------- WMMA fragment helpers (wave32, ISA 7.12.2 layouts) --------
// A (16x32 f16): lanes 0-15 row M=lane, halves 0-7 = K 0..7, halves 8-15 = K 16..23
//                lanes 16-31 row M=lane-16, K 8..15 and K 24..31.
// B (32x16) loaded symmetrically from transposed (NxK) storage.
__device__ static inline v16h load_frag(const h16* __restrict__ p, int ld,
                                        int row0, int k0, int lane) {
  int r  = row0 + (lane & 15);
  int kb = k0 + ((lane >> 4) << 3);
  const h16* b = p + r * ld + kb;
  v8h lo = *(const v8h*)(b);
  v8h hi = *(const v8h*)(b + 16);
  v16h a;
#pragma unroll
  for (int i = 0; i < 8; ++i) { a[i] = lo[i]; a[8 + i] = hi[i]; }
  return a;
}

__device__ static inline v8f wmma_f16(v16h a, v16h b, v8f c) {
  return __builtin_amdgcn_wmma_f32_16x16x32_f16(false, a, false, b, (short)0, c,
                                                false, false);
}

// Build TDM D# group0/group1 for a contiguous 16384-element fp32 (64KB) tile.
__device__ static inline void make_tdm_desc(unsigned long long ga,
                                            unsigned int ldsoff, v4u& g0,
                                            v8u& g1) {
  g0[0] = 1u;                                                     // count=1
  g0[1] = ldsoff;                                                 // lds_addr
  g0[2] = (unsigned int)(ga & 0xffffffffu);                       // addr[31:0]
  g0[3] = (unsigned int)((ga >> 32) & 0x01ffffffu) | (2u << 30);  // addr[56:32] | type=2
  g1[0] = 2u << 16;          // data_size = 4B
  g1[1] = 16384u << 16;      // tensor_dim0 = 16384 (low 16 bits)
  g1[2] = 1u << 16;          // tensor_dim0 hi = 0, tensor_dim1 = 1
  g1[3] = 16384u << 16;      // tile_dim0 = 16384
  g1[4] = 1u;                // tile_dim1 = 1
  g1[5] = 16384u;            // tensor_dim0_stride
  g1[6] = 0u;
  g1[7] = 0u;
}

// ---------------- fused per-read kernel ----------------
// LDS map (bytes):
//   [0      ,  65536) REUSE: E-fp32 (TDM dest) -> Zraw fp32 -> Treg(T/Hrev) + Sreg(S/VT)
//   [65536  ,  98304) Ef16 [l][d]        (reused as V after order 0)
//   [98304  , 131072) X0   [d][l] f16    (reused as V after order 1)
//   [131072 , 163840) X1   [d][l] f16 \  (reused together as 64KB fp32 OUT
//   [163840 , 196608) VA   [d][l] f16 /   staging block for the TDM store)
//   [196608 , ...   ) positions / mask / minpos
#define SM_BYTES (196608 + 1056)

__global__ __launch_bounds__(256) void hyena_read_kernel(
    const float* __restrict__ emb, const int* __restrict__ pos,
    const float* __restrict__ b_proj, const float* __restrict__ conv_w,
    const float* __restrict__ conv_b, const float* __restrict__ b1,
    const float* __restrict__ b2, const float* __restrict__ Bparam,
    const h16* __restrict__ wf, float* __restrict__ out) {
  __shared__ __align__(16) unsigned char smem[SM_BYTES];
  float* Zraw = (float*)(smem);                    // 64K (also E fp32 stage)
  h16*   Treg = (h16*)(smem);                      // 32K: T, then Hrev
  h16*   Sreg = (h16*)(smem + 32768);              // 32K: S, then VT
  h16*   Ef   = (h16*)(smem + 65536);
  h16*   X0b  = (h16*)(smem + 98304);
  h16*   X1b  = (h16*)(smem + 131072);
  h16*   VAb  = (h16*)(smem + 163840);
  float* OutL = (float*)(smem + 131072);           // 64K fp32 OUT staging
  int*   sp   = (int*)(smem + 196608);             // [128]
  float* smk  = (float*)(smem + 196608 + 512);     // [128]
  int*   smin = (int*)(smem + 196608 + 1024);

  const int tid  = threadIdx.x;
  const int lane = tid & 31;
  const int wave = tid >> 5;
  const int read = blockIdx.x;
  const unsigned int ldsbase = (unsigned int)(uintptr_t)(void*)smem;

  // positions, mask, min
  if (tid < RL) {
    int p = pos[read * RL + tid];
    sp[tid]  = p;
    smk[tid] = (p == -1) ? 0.0f : 1.0f;
  }

  // ---- TDM: DMA this read's 128x128 fp32 embedding tile into LDS ----
  if (wave == 0) {
    v4u g0; v8u g1;
    make_tdm_desc((unsigned long long)(uintptr_t)(emb + (size_t)read * RL * DIM),
                  ldsbase, g0, g1);
    asm volatile("tensor_load_to_lds %0, %1" :: "s"(g0), "s"(g1) : "memory");
    __builtin_amdgcn_s_wait_tensorcnt(0);
  }
  __syncthreads();

  if (tid == 0) {
    int mn = 0x7fffffff;
    for (int i = 0; i < RL; ++i)
      if (sp[i] != -1 && sp[i] < mn) mn = sp[i];
    *smin = (mn == 0x7fffffff) ? 0 : mn;
  }
  // convert E fp32 (staged in REUSE) -> f16 [l][d]
  for (int idx = tid; idx < RL * DIM; idx += 256) Ef[idx] = (h16)Zraw[idx];
  __syncthreads();

  // ---- stage 1: Z = E @ Wproj + b, causal depthwise conv(K=3), split ----
  for (int chunk = 0; chunk < 3; ++chunk) {
    const h16* Bw = wf + WS_WPROJT + (size_t)chunk * 128 * 128;
    for (int t = wave; t < 64; t += 8) {
      int m0 = (t >> 3) << 4, n0 = (t & 7) << 4;
      v8f acc = {};
#pragma unroll
      for (int k0 = 0; k0 < 128; k0 += 32) {
        v16h a = load_frag(Ef, 128, m0, k0, lane);
        v16h b = load_frag(Bw, 128, n0, k0, lane);
        acc = wmma_f16(a, b, acc);
      }
      int n  = n0 + (lane & 15);
      int mb = m0 + ((lane >> 4) << 3);
      float bias = b_proj[chunk * 128 + n];
#pragma unroll
      for (int j = 0; j < 8; ++j)
        Zraw[n * 128 + mb + j] = (acc[j] + bias) * smk[mb + j]; // [c][l]
    }
    __syncthreads();
    h16* dst = (chunk == 0) ? X0b : ((chunk == 1) ? X1b : VAb);
    for (int idx = tid; idx < 128 * 128; idx += 256) {
      int c = idx >> 7, l = idx & 127;
      int cg = chunk * 128 + c;
      float z2 = Zraw[c * 128 + l];
      float z1 = (l >= 1) ? Zraw[c * 128 + l - 1] : 0.0f;
      float z0 = (l >= 2) ? Zraw[c * 128 + l - 2] : 0.0f;
      float r = (z0 * conv_w[cg * 3] + z1 * conv_w[cg * 3 + 1] +
                 z2 * conv_w[cg * 3 + 2] + conv_b[cg]) * smk[l];
      dst[idx] = (h16)r;                                        // [d][l]
    }
    __syncthreads();
  }

  // ---- stage 2: RoPE(E) -> T (f16, [l][d]) into REUSE[0:32K] ----
  {
    int mn = *smin;
    for (int idx = tid; idx < RL * 64; idx += 256) {
      int l = idx >> 6, p = idx & 63;
      float adj = fmaxf((float)(sp[l] - mn), 0.0f);
      float ang = adj * __expf(-0.1439115683f * (float)p); // 10000^(-2p/128)
      float sn, cs;
      __sincosf(ang, &sn, &cs);
      float x1 = (float)Ef[l * 128 + 2 * p];
      float x2 = (float)Ef[l * 128 + 2 * p + 1];
      Treg[l * 128 + 2 * p]     = (h16)(x1 * cs - x2 * sn);
      Treg[l * 128 + 2 * p + 1] = (h16)(x1 * sn + x2 * cs);
    }
  }
  __syncthreads();

  // ---- stage 3: S = sin(T @ W1 + b1) -> Sreg (f16, [l][d]) ----
  for (int t = wave; t < 64; t += 8) {
    int m0 = (t >> 3) << 4, n0 = (t & 7) << 4;
    v8f acc = {};
#pragma unroll
    for (int k0 = 0; k0 < 128; k0 += 32) {
      v16h a = load_frag(Treg, 128, m0, k0, lane);
      v16h b = load_frag(wf + WS_W1T, 128, n0, k0, lane);
      acc = wmma_f16(a, b, acc);
    }
    int n  = n0 + (lane & 15);
    int mb = m0 + ((lane >> 4) << 3);
    float bias = b1[n];
#pragma unroll
    for (int j = 0; j < 8; ++j)
      Sreg[(mb + j) * 128 + n] = (h16)__sinf(acc[j] + bias);
  }
  __syncthreads();

  // ---- stage 4: two Hyena orders ----
  const h16* Vcur = VAb;
  for (int iord = 0; iord < 2; ++iord) {
    // Hrev[d][i] = h_i[d][127-i] = (S @ W2_i + b2_i)[127-i][d]  -> Treg region.
    // Reversed storage makes the causal-conv dot fully contiguous.
    const h16* Bw = wf + WS_W2T + (size_t)iord * 128 * 128;
    for (int t = wave; t < 64; t += 8) {
      int m0 = (t >> 3) << 4, n0 = (t & 7) << 4;
      v8f acc = {};
#pragma unroll
      for (int k0 = 0; k0 < 128; k0 += 32) {
        v16h a = load_frag(Sreg, 128, m0, k0, lane);
        v16h b = load_frag(Bw, 128, n0, k0, lane);
        acc = wmma_f16(a, b, acc);
      }
      int n  = n0 + (lane & 15);
      int mb = m0 + ((lane >> 4) << 3);
      float bias = b2[iord * 128 + n];
      v8h hv;
#pragma unroll
      for (int j = 0; j < 8; ++j) hv[j] = (h16)(acc[7 - j] + bias);
      *(v8h*)(Treg + n * 128 + (120 - mb)) = hv;   // reversed, 16B aligned
    }
    __syncthreads();

    // v <- x_i * (v*B_i + causal_conv(v, h_i)); both dot operands contiguous.
    const h16* Xb = (iord == 0) ? X0b : X1b;
    h16* VN = (iord == 0) ? Ef : X0b;              // reuse dead buffers
    for (int idx = tid; idx < 128 * 128; idx += 256) {
      int d = idx >> 7, l = idx & 127;
      const h16* vr = Vcur + d * 128;
      const h16* hr = Treg + d * 128 + (127 - l);  // hr[k] == h[l-k]
      float acc = 0.0f;
      int e = l + 1, k = 0;
      for (; k + 4 <= e; k += 4) {
        acc += (float)vr[k]     * (float)hr[k];
        acc += (float)vr[k + 1] * (float)hr[k + 1];
        acc += (float)vr[k + 2] * (float)hr[k + 2];
        acc += (float)vr[k + 3] * (float)hr[k + 3];
      }
      for (; k < e; ++k) acc += (float)vr[k] * (float)hr[k];
      float nv = (float)Xb[idx] *
                 ((float)vr[l] * Bparam[iord * 128 + d] + acc * smk[l]);
      VN[idx] = (h16)nv;
    }
    __syncthreads();
    Vcur = VN;
  }

  // ---- stage 5: out = V^T @ Wout, staged in LDS, DMA'd out via TDM ----
  for (int idx = tid; idx < 128 * 128; idx += 256) {  // transpose [d][l]->[l][d]
    int d = idx >> 7, l = idx & 127;
    Sreg[l * 128 + d] = Vcur[idx];
  }
  __syncthreads();
  for (int t = wave; t < 64; t += 8) {
    int m0 = (t >> 3) << 4, n0 = (t & 7) << 4;
    v8f acc = {};
#pragma unroll
    for (int k0 = 0; k0 < 128; k0 += 32) {
      v16h a = load_frag(Sreg, 128, m0, k0, lane);
      v16h b = load_frag(wf + WS_WOUTT, 128, n0, k0, lane);
      acc = wmma_f16(a, b, acc);
    }
    int n  = n0 + (lane & 15);
    int mb = m0 + ((lane >> 4) << 3);
#pragma unroll
    for (int j = 0; j < 8; ++j)
      OutL[(mb + j) * 128 + n] = acc[j];           // fp32 [l][d] staging
  }
  __syncthreads();

  if (wave == 0) {
    v4u g0; v8u g1;
    make_tdm_desc((unsigned long long)(uintptr_t)(out + (size_t)read * RL * DIM),
                  ldsbase + 131072u, g0, g1);
    asm volatile("tensor_store_from_lds %0, %1" :: "s"(g0), "s"(g1) : "memory");
    __builtin_amdgcn_s_wait_tensorcnt(0);
  }
}

// ---------------- host entry ----------------
extern "C" void kernel_launch(void* const* d_in, const int* in_sizes, int n_in,
                              void* d_out, int out_size, void* d_ws,
                              size_t ws_size, hipStream_t stream) {
  const float* emb    = (const float*)d_in[0];
  const int*   pos    = (const int*)d_in[1];
  const float* Wproj  = (const float*)d_in[2];
  const float* b_proj = (const float*)d_in[3];
  const float* conv_w = (const float*)d_in[4];
  const float* conv_b = (const float*)d_in[5];
  const float* W1     = (const float*)d_in[6];
  const float* b1     = (const float*)d_in[7];
  const float* W2     = (const float*)d_in[8];
  const float* b2     = (const float*)d_in[9];
  const float* Bp     = (const float*)d_in[10];
  const float* Wout   = (const float*)d_in[11];
  h16* wf = (h16*)d_ws;
  float* outp = (float*)d_out;

  int nReads = in_sizes[0] / (RL * DIM);  // (B*SEQ)/READ_LEN = 512

  prep_weights<<<(WS_TOTAL + 255) / 256, 256, 0, stream>>>(Wproj, W1, W2, Wout, wf);
  hyena_read_kernel<<<nReads, 256, 0, stream>>>(emb, pos, b_proj, conv_w, conv_b,
                                                b1, b2, Bp, wf, outp);
}
